// ModulatedDeformConv2dDeterministic_12996571038287
// MI455X (gfx1250) — matrix operand order repaired and checked by reading, not compiled, not run on hardware
//
#include <hip/hip_runtime.h>
#include <hip/hip_bf16.h>

// ---------------------------------------------------------------------------
// Modulated Deformable Conv2d (DCNv2) forward for gfx1250 (MI455X, wave32).
//
//   out[b,o,p] = bias[o] + sum_{c,k} W[o, c*9+k] * cols[c*9+k, (b,p)]
//   cols = bilinear-sampled, mask-modulated im2col (K = 3x3, pad=1, stride=1)
//
// Shapes: B=4, Cin=Cout=256, H=W=Ho=Wo=64, Kdim = 256*9 = 2304, Npos = 4096/batch.
//
// Strategy (see analysis): f16 storage + f32-accumulate WMMA
// (v_wmma_f32_16x16x32_f16), per-batch cols (18.9 MB, L2-resident),
// producer-side swizzling into exact WMMA fragment layouts so the GEMM loads
// each lane's v16h fragment as one contiguous 32B global load.
// ---------------------------------------------------------------------------

typedef __attribute__((ext_vector_type(16))) _Float16 v16h;
typedef __attribute__((ext_vector_type(8)))  float    v8f;

#define CIN   256
#define COUT  256
#define HH    64
#define WW    64
#define KK    9            // 3x3
#define KDIM  (CIN * KK)   // 2304
#define NPOS  (HH * WW)    // 4096 spatial positions per batch
#define KT    (KDIM / 32)  // 72 K-tiles of 32
#define MT16  (COUT / 16)  // 16 M fragments of 16 rows
#define NT16  (NPOS / 16)  // 256 N fragments of 16 cols (per batch)

// ---------------------------------------------------------------------------
// Kernel 0: convert weight f32 [COUT][KDIM] into f16 WMMA A-fragments.
// A tile (tm, tk) covers rows m = tm*16..+15, k = tk*32..+31.
// 16-bit A 16x32, contiguous k order (not the ISA table order): lane holds row m = lane&15; half h maps to
//   koff = h + ((lane>>4)<<4)
// Stored: wp[((tm*KT + tk)*32 + lane)*16 + h]  -> 32B contiguous per lane.
// ---------------------------------------------------------------------------
__global__ void mdcn_pack_weight(const float* __restrict__ w,
                                 _Float16* __restrict__ wp) {
    int gid = blockIdx.x * blockDim.x + threadIdx.x;     // (tile, lane)
    if (gid >= MT16 * KT * 32) return;
    int lane   = gid & 31;
    int tileid = gid >> 5;
    int tm     = tileid / KT;
    int tk     = tileid % KT;
    int m      = tm * 16 + (lane & 15);
    const float* wrow = w + (size_t)m * KDIM + tk * 32;
    _Float16* dst = wp + ((size_t)tileid * 32 + lane) * 16;
#pragma unroll
    for (int h = 0; h < 16; ++h) {
        int koff = h + ((lane >> 4) << 4);
        dst[h] = (_Float16)wrow[koff];
    }
}

// ---------------------------------------------------------------------------
// Kernel 1: deformable im2col for one batch -> swizzled f16 B-fragments.
// B tile (tk, tn) covers k-rows rk = 0..31, cols n = tn*16..+15.
// 16-bit B 32x16, contiguous k order as in A (not the ISA table order): lane = (rk&16) | (n&15), half h = rk&15.
// Stored: cols[((tk*NT16 + tn)*32 + lane)*16 + h].
// One thread per (k, p); loops over the 256 input channels (4 gathers + fma
// each); mask and corner validity folded into the 4 bilinear weights.
// ---------------------------------------------------------------------------
__global__ void mdcn_im2col(const float* __restrict__ x,
                            const float* __restrict__ offset,
                            const float* __restrict__ mask,
                            _Float16* __restrict__ cols, int b) {
    int gid = blockIdx.x * blockDim.x + threadIdx.x;
    if (gid >= KK * NPOS) return;
    int k  = gid / NPOS;          // adjacent threads share k -> coalesced
    int p  = gid % NPOS;
    int ho = p >> 6, wo = p & 63;
    int kh = k / 3,  kw = k % 3;

    float dy = offset[((size_t)(b * 2 * KK) + (size_t)(k * 2 + 0)) * NPOS + p];
    float dx = offset[((size_t)(b * 2 * KK) + (size_t)(k * 2 + 1)) * NPOS + p];
    float mk = mask[((size_t)(b * KK) + k) * NPOS + p];

    float sy = (float)(ho - 1 + kh) + dy;
    float sx = (float)(wo - 1 + kw) + dx;
    float y0f = floorf(sy), x0f = floorf(sx);
    float wy = sy - y0f,    wx = sx - x0f;
    int y0 = (int)y0f,      x0 = (int)x0f;
    int y1 = y0 + 1,        x1 = x0 + 1;

    bool vy0 = (y0 >= 0) & (y0 < HH), vy1 = (y1 >= 0) & (y1 < HH);
    bool vx0 = (x0 >= 0) & (x0 < WW), vx1 = (x1 >= 0) & (x1 < WW);
    int cy0 = min(max(y0, 0), HH - 1), cy1 = min(max(y1, 0), HH - 1);
    int cx0 = min(max(x0, 0), WW - 1), cx1 = min(max(x1, 0), WW - 1);

    // mask * validity folded into the 4 corner weights
    float w00 = (1.f - wy) * (1.f - wx) * ((vy0 & vx0) ? mk : 0.f);
    float w01 = (1.f - wy) * wx         * ((vy0 & vx1) ? mk : 0.f);
    float w10 = wy * (1.f - wx)         * ((vy1 & vx0) ? mk : 0.f);
    float w11 = wy * wx                 * ((vy1 & vx1) ? mk : 0.f);

    int i00 = cy0 * WW + cx0, i01 = cy0 * WW + cx1;
    int i10 = cy1 * WW + cx0, i11 = cy1 * WW + cx1;

    const float* xb = x + (size_t)b * CIN * NPOS;
    int tn = p >> 4;
    int cn = p & 15;

    for (int c = 0; c < CIN; ++c) {
        const float* xc = xb + (size_t)c * NPOS;
        float v = fmaf(w00, xc[i00],
                  fmaf(w01, xc[i01],
                  fmaf(w10, xc[i10],
                        w11 * xc[i11])));
        int kk = c * KK + k;               // matches weight's (c,kh,kw) flatten
        int tk = kk >> 5;
        int rk = kk & 31;
        int lane = (rk & 16) | cn;
        int h    = rk & 15;
        cols[(((size_t)tk * NT16 + tn) * 32 + lane) * 16 + h] = (_Float16)v;
    }
}

// ---------------------------------------------------------------------------
// Kernel 2: WMMA GEMM  D[m,n] = sum_k Wp[m,k] * cols[k,n]  + bias[m]
// Per wave: 32x32 output tile = 2x2 f32 accumulators; 72 K-steps of 32;
// 4x v_wmma_f32_16x16x32_f16 per step. Fragments are contiguous 32B/lane.
// Epilogue scatters per the f32 16x16 C/D layout (lane<16: M=r, lane>=16: M=8+r).
// ---------------------------------------------------------------------------
__global__ void mdcn_wmma_gemm(const _Float16* __restrict__ wp,
                               const _Float16* __restrict__ cols,
                               const float* __restrict__ bias,
                               float* __restrict__ out, int b) {
    int lane = threadIdx.x & 31;
    int wave = threadIdx.x >> 5;
    int gw   = blockIdx.x * 4 + wave;       // 0..1023
    int tm2  = gw >> 7;                     // 0..7   (32-row M block)
    int tn2  = gw & 127;                    // 0..127 (32-col N block)
    int tmA0 = tm2 * 2, tmA1 = tm2 * 2 + 1; // 16-row A fragment indices
    int tnB0 = tn2 * 2, tnB1 = tn2 * 2 + 1; // 16-col B fragment indices

    v8f acc00 = {}, acc01 = {}, acc10 = {}, acc11 = {};

    const _Float16* pa0 = wp   + ((size_t)(tmA0 * KT) * 32 + lane) * 16;
    const _Float16* pa1 = wp   + ((size_t)(tmA1 * KT) * 32 + lane) * 16;
    const _Float16* pb0 = cols + ((size_t)tnB0 * 32 + lane) * 16;
    const _Float16* pb1 = cols + ((size_t)tnB1 * 32 + lane) * 16;
    const size_t strideA = (size_t)32 * 16;             // next tk in wp
    const size_t strideB = (size_t)NT16 * 32 * 16;      // next tk in cols

#pragma unroll 2
    for (int tk = 0; tk < KT; ++tk) {
        v16h a0 = *(const v16h*)(pa0 + (size_t)tk * strideA);
        v16h a1 = *(const v16h*)(pa1 + (size_t)tk * strideA);
        v16h b0 = *(const v16h*)(pb0 + (size_t)tk * strideB);
        v16h b1 = *(const v16h*)(pb1 + (size_t)tk * strideB);
        acc00 = __builtin_amdgcn_wmma_f32_16x16x32_f16(false, a0, false, b0,
                                                       (short)0, acc00, false, false);
        acc01 = __builtin_amdgcn_wmma_f32_16x16x32_f16(false, a0, false, b1,
                                                       (short)0, acc01, false, false);
        acc10 = __builtin_amdgcn_wmma_f32_16x16x32_f16(false, a1, false, b0,
                                                       (short)0, acc10, false, false);
        acc11 = __builtin_amdgcn_wmma_f32_16x16x32_f16(false, a1, false, b1,
                                                       (short)0, acc11, false, false);
    }

    // Epilogue: C/D 16x16 f32 layout -> out[b][m][p], p = n (per-batch block)
    int n_local = lane & 15;
    int m_hi    = (lane >> 4) << 3;         // 0 or 8
    float* ob = out + (size_t)b * COUT * NPOS;

#pragma unroll
    for (int i = 0; i < 2; ++i) {
        int m_base = tm2 * 32 + i * 16 + m_hi;
        const v8f* accs0 = (i == 0) ? &acc00 : &acc10;
        const v8f* accs1 = (i == 0) ? &acc01 : &acc11;
#pragma unroll
        for (int r = 0; r < 8; ++r) {
            int m = m_base + r;
            float bi = bias[m];
            ob[(size_t)m * NPOS + (tn2 * 32 + n_local)]      = (*accs0)[r] + bi;
            ob[(size_t)m * NPOS + (tn2 * 32 + 16 + n_local)] = (*accs1)[r] + bi;
        }
    }
}

// ---------------------------------------------------------------------------
// Launch: pack weight once, then per batch: im2col -> WMMA GEMM (cols for one
// batch = 18.9 MB, stays resident in the 192 MB L2 between phases).
// Workspace layout: [0, 1.18MB) packed weight f16 ; then per-batch cols f16.
// ---------------------------------------------------------------------------
extern "C" void kernel_launch(void* const* d_in, const int* in_sizes, int n_in,
                              void* d_out, int out_size, void* d_ws, size_t ws_size,
                              hipStream_t stream) {
    const float* x      = (const float*)d_in[0];
    const float* offset = (const float*)d_in[1];
    const float* mask   = (const float*)d_in[2];
    const float* weight = (const float*)d_in[3];   // [COUT][CIN][3][3] == [COUT][KDIM]
    const float* bias   = (const float*)d_in[4];
    float* out = (float*)d_out;

    char* ws = (char*)d_ws;
    _Float16* wp   = (_Float16*)ws;                                   // 1.18 MB
    _Float16* cols = (_Float16*)(ws + (size_t)MT16 * KT * 32 * 16 * 2); // 18.9 MB

    // Phase 0: weight -> f16 A fragments (16*72 tiles * 32 lanes threads)
    {
        int total = MT16 * KT * 32;                  // 36864
        mdcn_pack_weight<<<(total + 255) / 256, 256, 0, stream>>>(weight, wp);
    }

    const int B = 4;
    for (int b = 0; b < B; ++b) {
        // Phase 1: deformable im2col (one thread per (k, p), loops channels)
        {
            int total = KK * NPOS;                   // 36864
            mdcn_im2col<<<(total + 255) / 256, 256, 0, stream>>>(x, offset, mask,
                                                                 cols, b);
        }
        // Phase 2: WMMA GEMM, 1024 waves = 256 blocks of 4 waves (128 thr)
        {
            mdcn_wmma_gemm<<<256, 128, 0, stream>>>(wp, cols, bias, out, b);
        }
    }
}
